// ChunkedMultiHeadCardPassingLayer_30863634989718
// MI455X (gfx1250) — matrix operand order repaired and checked by reading, not compiled, not run on hardware
//
#include <hip/hip_runtime.h>

#define DEV __device__ __forceinline__

// Problem constants (hardcoded per reference)
constexpr int Bn = 4, Tn = 4096, Cn = 1024, Hn = 16, CSn = 128, Dn = 64, NCn = 32;
constexpr float EPSf = 1e-5f;

typedef __attribute__((ext_vector_type(16))) __bf16 bf16x16;
typedef __attribute__((ext_vector_type(8)))  float  f32x8;

union FragU { unsigned int u[8]; bf16x16 v; };

DEV unsigned short f2bf(float f) {
  unsigned int u = __float_as_uint(f);
  u += 0x7FFFu + ((u >> 16) & 1u);       // round-to-nearest-even
  return (unsigned short)(u >> 16);
}

DEV float gelu_exact(float x) { return 0.5f * x * (1.0f + erff(x * 0.70710678118654752f)); }

DEV f32x8 wmma_bf16(bf16x16 a, bf16x16 b, f32x8 c) {
  return __builtin_amdgcn_wmma_f32_16x16x32_bf16(false, a, false, b, (short)0, c, false, false);
}

// --- CDNA5 async global->LDS copy (ASYNCcnt path, no VGPR staging) ----------
// Generic LDS pointer's low 32 bits ARE the LDS byte offset.
DEV void async_ld16(const unsigned short* gsrc, unsigned short* ldst) {
  unsigned loff = (unsigned)(unsigned long long)(size_t)ldst;
  asm volatile("global_load_async_to_lds_b128 %0, %1, off"
               :: "v"(loff), "v"(gsrc) : "memory");
}
#define WAIT_ASYNC(n) asm volatile("s_wait_asynccnt " #n ::: "memory")

// A fragment: 16x32 bf16, LDS tile row-major [m][k].
// lanes 0-15: M=lane, K in {0..7, 16..23}; lanes 16-31: M=lane-16, K in {8..15, 24..31}
DEV bf16x16 load_a_frag(const unsigned short* As, int stride, int mBase, int lane) {
  int m  = mBase + (lane & 15);
  int kb = (lane >= 16) ? 8 : 0;
  FragU f;
#pragma unroll
  for (int p = 0; p < 8; ++p) {
    int k = (p < 4) ? (kb + 2 * p) : (kb + 8 + 2 * p);
    f.u[p] = *(const unsigned int*)(As + m * stride + k);
  }
  return f.v;
}

// B fragment: 32x16 bf16, LDS tile stored [n][k] (weights pre-transposed in HBM).
// lanes 0-15: N=lane, K in {0..7, 16..23}; lanes 16-31: N=lane-16, K in {8..15, 24..31}
DEV bf16x16 load_b_frag(const unsigned short* Bs, int stride, int nBase, int lane) {
  int n  = nBase + (lane & 15);
  int kb = (lane >= 16) ? 8 : 0;
  FragU f;
#pragma unroll
  for (int p = 0; p < 8; ++p)
    f.u[p] = *(const unsigned int*)(Bs + n * stride + ((p < 4) ? (kb + 2 * p) : (kb + 8 + 2 * p)));
  return f.v;
}

// ---------------------------------------------------------------------------
// 64x64-tile bf16 WMMA GEMM, double-buffered async global->LDS pipeline.
// B operand is PRE-TRANSPOSED: Bt[n*K + k]. 256 thr = 8 waves, each wave owns
// 2 of 16 16x16 tiles (same column -> shared B fragment).
// ---------------------------------------------------------------------------
enum { EPI_F32 = 0, EPI_GELU_BF16 = 1, EPI_HEADOUT_BF16 = 2 };
constexpr int STRL = 40;  // LDS row stride (ushorts): 80B keeps 16B chunks aligned,
                          // 20-dword stride -> conflict-free 16-lane fragment reads

template <int EPI>
__global__ __launch_bounds__(256) void gemm_bf16_kernel(
    const unsigned short* __restrict__ A, const unsigned short* __restrict__ Bt,
    const float* __restrict__ bias, void* __restrict__ outp, int M, int N, int K) {
  __shared__ __align__(16) unsigned short As[2][64 * STRL];
  __shared__ __align__(16) unsigned short Bs[2][64 * STRL];
  const int tid = threadIdx.x, wave = tid >> 5, lane = tid & 31;
  const int rowBase = blockIdx.y * 64, colBase = blockIdx.x * 64;
  const int tn = wave & 3, tm0 = wave >> 2, tm1 = tm0 + 2;
  // per-thread 16B chunk: row mrow, ushort offset chk within the 64x32 tile
  const int mrow = tid >> 2, chk = (tid & 3) * 8;
  const size_t aOff = (size_t)(rowBase + mrow) * K + chk;
  const size_t bOff = (size_t)(colBase + mrow) * K + chk;
  const int lOff = mrow * STRL + chk;

  f32x8 acc0 = {}, acc1 = {};
  const int S = K >> 5;
  async_ld16(A + aOff, &As[0][lOff]);
  async_ld16(Bt + bOff, &Bs[0][lOff]);
  for (int s = 0; s < S; ++s) {
    const int cur = s & 1;
    if (s + 1 < S) {
      const int k1 = (s + 1) << 5;
      async_ld16(A + aOff + k1, &As[1 - cur][lOff]);
      async_ld16(Bt + bOff + k1, &Bs[1 - cur][lOff]);
      WAIT_ASYNC(2);   // own loads for buf[cur] complete (in-order retire)
    } else {
      WAIT_ASYNC(0);
    }
    __syncthreads();   // everyone's buf[cur] data visible
    bf16x16 bf = load_b_frag(Bs[cur], STRL, tn * 16, lane);
    bf16x16 a0 = load_a_frag(As[cur], STRL, tm0 * 16, lane);
    bf16x16 a1 = load_a_frag(As[cur], STRL, tm1 * 16, lane);
    acc0 = wmma_bf16(a0, bf, acc0);
    acc1 = wmma_bf16(a1, bf, acc1);
    __syncthreads();   // done reading before buf[cur] is overwritten
  }
  const int col  = colBase + tn * 16 + (lane & 15);
  const float bv = bias[col];
  const int moff = (lane >= 16) ? 8 : 0;
#pragma unroll
  for (int r = 0; r < 8; ++r) {
    int row0 = rowBase + tm0 * 16 + r + moff;
    int row1 = rowBase + tm1 * 16 + r + moff;
    float v0 = acc0[r] + bv, v1 = acc1[r] + bv;
    if (EPI == EPI_F32) {
      float* O = (float*)outp;
      O[(size_t)row0 * N + col] = v0;
      O[(size_t)row1 * N + col] = v1;
    } else if (EPI == EPI_GELU_BF16) {
      unsigned short* O = (unsigned short*)outp;
      O[(size_t)row0 * N + col] = f2bf(gelu_exact(v0));
      O[(size_t)row1 * N + col] = f2bf(gelu_exact(v1));
    } else {  // head rows (b,h,nc,cs) -> (B,T,C) with c = h*D + col
      unsigned short* O = (unsigned short*)outp;
      int b0 = row0 >> 16, h0 = (row0 >> 12) & 15, t0 = row0 & 4095;
      int b1 = row1 >> 16, h1 = (row1 >> 12) & 15, t1 = row1 & 4095;
      O[((size_t)(b0 * Tn + t0)) * Cn + h0 * Dn + col] = f2bf(v0);
      O[((size_t)(b1 * Tn + t1)) * Cn + h1 * Dn + col] = f2bf(v1);
    }
  }
}

// ---------------------------------------------------------------------------
// Fused dual GEMM: gated = sigmoid(x@Wg + bg) * (x@Wm + bm). One async A tile
// feeds both (pre-transposed) B matrices.
// ---------------------------------------------------------------------------
__global__ __launch_bounds__(256) void gemm_dual_gated_kernel(
    const unsigned short* __restrict__ A, const unsigned short* __restrict__ Wmt,
    const unsigned short* __restrict__ Wgt, const float* __restrict__ bm,
    const float* __restrict__ bg, float* __restrict__ gated, int M, int N, int K) {
  __shared__ __align__(16) unsigned short As[2][64 * STRL];
  __shared__ __align__(16) unsigned short Bm[2][64 * STRL];
  __shared__ __align__(16) unsigned short Bg[2][64 * STRL];
  const int tid = threadIdx.x, wave = tid >> 5, lane = tid & 31;
  const int rowBase = blockIdx.y * 64, colBase = blockIdx.x * 64;
  const int tn = wave & 3, tm0 = wave >> 2, tm1 = tm0 + 2;
  const int mrow = tid >> 2, chk = (tid & 3) * 8;
  const size_t aOff = (size_t)(rowBase + mrow) * K + chk;
  const size_t bOff = (size_t)(colBase + mrow) * K + chk;
  const int lOff = mrow * STRL + chk;

  f32x8 am0 = {}, am1 = {}, ag0 = {}, ag1 = {};
  const int S = K >> 5;
  async_ld16(A + aOff, &As[0][lOff]);
  async_ld16(Wmt + bOff, &Bm[0][lOff]);
  async_ld16(Wgt + bOff, &Bg[0][lOff]);
  for (int s = 0; s < S; ++s) {
    const int cur = s & 1;
    if (s + 1 < S) {
      const int k1 = (s + 1) << 5;
      async_ld16(A + aOff + k1, &As[1 - cur][lOff]);
      async_ld16(Wmt + bOff + k1, &Bm[1 - cur][lOff]);
      async_ld16(Wgt + bOff + k1, &Bg[1 - cur][lOff]);
      WAIT_ASYNC(3);
    } else {
      WAIT_ASYNC(0);
    }
    __syncthreads();
    bf16x16 bMf = load_b_frag(Bm[cur], STRL, tn * 16, lane);
    bf16x16 bGf = load_b_frag(Bg[cur], STRL, tn * 16, lane);
    bf16x16 a0  = load_a_frag(As[cur], STRL, tm0 * 16, lane);
    bf16x16 a1  = load_a_frag(As[cur], STRL, tm1 * 16, lane);
    am0 = wmma_bf16(a0, bMf, am0);
    am1 = wmma_bf16(a1, bMf, am1);
    ag0 = wmma_bf16(a0, bGf, ag0);
    ag1 = wmma_bf16(a1, bGf, ag1);
    __syncthreads();
  }
  const int col = colBase + tn * 16 + (lane & 15);
  const float bmv = bm[col], bgv = bg[col];
  const int moff = (lane >= 16) ? 8 : 0;
#pragma unroll
  for (int r = 0; r < 8; ++r) {
    int row0 = rowBase + tm0 * 16 + r + moff;
    int row1 = rowBase + tm1 * 16 + r + moff;
    float m0 = am0[r] + bmv, g0 = ag0[r] + bgv;
    float m1 = am1[r] + bmv, g1 = ag1[r] + bgv;
    gated[(size_t)row0 * N + col] = m0 * (1.0f / (1.0f + expf(-g0)));
    gated[(size_t)row1 * N + col] = m1 * (1.0f / (1.0f + expf(-g1)));
  }
}

// ---------------------------------------------------------------------------
// f32 [K][N] weight -> bf16 [N][K] transposed (LDS-tiled, coalesced both ways)
__global__ __launch_bounds__(256) void transpose_convert_kernel(
    const float* __restrict__ W, unsigned short* __restrict__ Wt, int K, int N) {
  __shared__ float tile[32][33];
  const int n0 = blockIdx.x * 32, k0 = blockIdx.y * 32;
  const int tx = threadIdx.x & 31, ty = threadIdx.x >> 5;  // ty in 0..7
#pragma unroll
  for (int i = 0; i < 32; i += 8)
    tile[ty + i][tx] = W[(size_t)(k0 + ty + i) * N + n0 + tx];
  __syncthreads();
#pragma unroll
  for (int i = 0; i < 32; i += 8)
    Wt[(size_t)(n0 + ty + i) * K + k0 + tx] = f2bf(tile[tx][ty + i]);
}

// packed f32 -> bf16 (2 at a time)
__global__ void f32_to_bf16_kernel(const float2* __restrict__ in,
                                   unsigned int* __restrict__ out, size_t n2) {
  size_t i = (size_t)blockIdx.x * blockDim.x + threadIdx.x;
  size_t st = (size_t)gridDim.x * blockDim.x;
  for (; i < n2; i += st) {
    float2 v = in[i];
    out[i] = (unsigned)f2bf(v.x) | ((unsigned)f2bf(v.y) << 16);
  }
}

// chunk_sums[b,nc,c] = sum_cs gated[b, nc*CS+cs, c]
__global__ __launch_bounds__(256) void chunk_sums_kernel(
    const float* __restrict__ gated, float* __restrict__ csums) {
  int nc = blockIdx.x, b = blockIdx.y;
  for (int c = threadIdx.x; c < Cn; c += 256) {
    float s = 0.0f;
    for (int i = 0; i < CSn; ++i)
      s += gated[((size_t)(b * Tn + nc * CSn + i)) * Cn + c];
    csums[((size_t)(b * NCn + nc)) * Cn + c] = s;
  }
}

// Exclusive carry across chunks + LN over D -> norm_carries[b,nc,c]
__global__ __launch_bounds__(64) void carry_ln_kernel(
    const float* __restrict__ csums, const float* __restrict__ g,
    const float* __restrict__ be, float* __restrict__ ncarr) {
  __shared__ float red[64];
  const int h = blockIdx.x, b = blockIdx.y, d = threadIdx.x;
  float carry = 0.0f;
  const float gd = g[d], bd = be[d];
  for (int nc = 0; nc < NCn; ++nc) {
    red[d] = carry; __syncthreads();
    for (int s = 32; s > 0; s >>= 1) { if (d < s) red[d] += red[d + s]; __syncthreads(); }
    float mean = red[0] * (1.0f / 64.0f); __syncthreads();
    float diff = carry - mean;
    red[d] = diff * diff; __syncthreads();
    for (int s = 32; s > 0; s >>= 1) { if (d < s) red[d] += red[d + s]; __syncthreads(); }
    float inv = rsqrtf(red[0] * (1.0f / 64.0f) + EPSf); __syncthreads();
    size_t idx = ((size_t)(b * NCn + nc)) * Cn + h * Dn + d;
    ncarr[idx] = diff * inv * gd + bd;
    carry += csums[idx];
  }
}

// cards = LN(norm_carry + exclusive-prefix(gated)); emit combined=[x|cards] bf16.
// One wave per (b,h,nc); lane owns d and d+32; LN via wave32 shuffles.
__global__ __launch_bounds__(32) void cards_combined_kernel(
    const float* __restrict__ gated, const float* __restrict__ ncarr,
    const float* __restrict__ xf, const float* __restrict__ cg,
    const float* __restrict__ cb, unsigned short* __restrict__ combined) {
  const int nc = blockIdx.x, h = blockIdx.y, b = blockIdx.z, lane = threadIdx.x;
  const int d0 = lane, d1 = lane + 32;
  size_t cbase = ((size_t)(b * NCn + nc)) * Cn + h * Dn;
  const float nv0 = ncarr[cbase + d0], nv1 = ncarr[cbase + d1];
  const float g0 = cg[d0], g1 = cg[d1], b0 = cb[d0], b1 = cb[d1];
  float run0 = 0.0f, run1 = 0.0f;
  const size_t rowBase = ((size_t)((b * Hn + h) * NCn + nc)) * CSn;
  for (int cs = 0; cs < CSn; ++cs) {
    size_t xrow = ((size_t)(b * Tn + nc * CSn + cs)) * Cn + h * Dn;
    float c0 = nv0 + run0, c1 = nv1 + run1;
    float s = c0 + c1;
    for (int m = 16; m > 0; m >>= 1) s += __shfl_xor(s, m, 32);
    float mean = s * (1.0f / 64.0f);
    float e0 = c0 - mean, e1 = c1 - mean;
    float v = e0 * e0 + e1 * e1;
    for (int m = 16; m > 0; m >>= 1) v += __shfl_xor(v, m, 32);
    float inv = rsqrtf(v * (1.0f / 64.0f) + EPSf);
    size_t crow = (rowBase + cs) * 128;
    combined[crow + d0]      = f2bf(xf[xrow + d0]);
    combined[crow + d1]      = f2bf(xf[xrow + d1]);
    combined[crow + 64 + d0] = f2bf(e0 * inv * g0 + b0);
    combined[crow + 64 + d1] = f2bf(e1 * inv * g1 + b1);
    run0 += gated[xrow + d0];
    run1 += gated[xrow + d1];
  }
}

// out = x + LN(proj_out) over C
__global__ __launch_bounds__(256) void ln_residual_kernel(
    const float* __restrict__ x, const float* __restrict__ y,
    const float* __restrict__ g, const float* __restrict__ be,
    float* __restrict__ out) {
  __shared__ float red[256];
  const int row = blockIdx.x, tid = threadIdx.x;
  const size_t base = (size_t)row * Cn;
  float v[4]; float s = 0.0f;
#pragma unroll
  for (int i = 0; i < 4; ++i) { v[i] = y[base + tid + 256 * i]; s += v[i]; }
  red[tid] = s; __syncthreads();
  for (int t = 128; t > 0; t >>= 1) { if (tid < t) red[tid] += red[tid + t]; __syncthreads(); }
  float mean = red[0] * (1.0f / 1024.0f); __syncthreads();
  float q = 0.0f;
#pragma unroll
  for (int i = 0; i < 4; ++i) { float d = v[i] - mean; q += d * d; }
  red[tid] = q; __syncthreads();
  for (int t = 128; t > 0; t >>= 1) { if (tid < t) red[tid] += red[tid + t]; __syncthreads(); }
  float inv = rsqrtf(red[0] * (1.0f / 1024.0f) + EPSf); __syncthreads();
#pragma unroll
  for (int i = 0; i < 4; ++i) {
    int c = tid + 256 * i;
    out[base + c] = x[base + c] + (v[i] - mean) * inv * g[c] + be[c];
  }
}

// ---------------------------------------------------------------------------
extern "C" void kernel_launch(void* const* d_in, const int* in_sizes, int n_in,
                              void* d_out, int out_size, void* d_ws, size_t ws_size,
                              hipStream_t stream) {
  const float* x      = (const float*)d_in[0];
  const float* mark_W = (const float*)d_in[1];
  const float* mark_b = (const float*)d_in[2];
  const float* gate_W = (const float*)d_in[3];
  const float* gate_b = (const float*)d_in[4];
  const float* carry_g = (const float*)d_in[5];
  const float* carry_b = (const float*)d_in[6];
  const float* card_g  = (const float*)d_in[7];
  const float* card_b  = (const float*)d_in[8];
  const float* W1     = (const float*)d_in[9];
  const float* b1     = (const float*)d_in[10];
  const float* W2     = (const float*)d_in[11];
  const float* b2     = (const float*)d_in[12];
  const float* proj_W = (const float*)d_in[13];
  const float* proj_b = (const float*)d_in[14];
  const float* ln_g   = (const float*)d_in[15];
  const float* ln_b   = (const float*)d_in[16];
  float* out = (float*)d_out;

  const size_t BTC = (size_t)Bn * Tn * Cn;          // 16,777,216
  const size_t MH  = (size_t)Bn * Hn * NCn * CSn;   // 262,144 head rows

  unsigned char* w = (unsigned char*)d_ws;
  size_t off = 0;
  auto take = [&](size_t bytes) -> void* {
    void* p = w + off; off += bytes; off = (off + 255) & ~(size_t)255; return p;
  };
  unsigned short* xb     = (unsigned short*)take(BTC * 2);
  unsigned short* markWt = (unsigned short*)take((size_t)Cn * Cn * 2);  // [N][K]
  unsigned short* gateWt = (unsigned short*)take((size_t)Cn * Cn * 2);
  unsigned short* projWt = (unsigned short*)take((size_t)Cn * Cn * 2);
  unsigned short* W1t    = (unsigned short*)take((size_t)128 * 128 * 2);
  unsigned short* W2t    = (unsigned short*)take((size_t)64 * 128 * 2);
  float* gated    = (float*)take(BTC * 4);
  float* csums    = (float*)take((size_t)Bn * NCn * Cn * 4);
  float* ncarr    = (float*)take((size_t)Bn * NCn * Cn * 4);
  unsigned short* combined = (unsigned short*)take(MH * 128 * 2);
  unsigned short* head_out = (unsigned short*)take(BTC * 2);
  unsigned short* hbuf = (unsigned short*)gated;   // alias: gated dead after cards kernel
  float* proj_out = (float*)combined;              // alias: combined dead after head GEMM1

  // 1) conversions: x -> bf16; weights -> bf16 transposed [N][K]
  f32_to_bf16_kernel<<<4096, 256, 0, stream>>>((const float2*)x, (unsigned int*)xb, BTC / 2);
  transpose_convert_kernel<<<dim3(32, 32), 256, 0, stream>>>(mark_W, markWt, Cn, Cn);
  transpose_convert_kernel<<<dim3(32, 32), 256, 0, stream>>>(gate_W, gateWt, Cn, Cn);
  transpose_convert_kernel<<<dim3(32, 32), 256, 0, stream>>>(proj_W, projWt, Cn, Cn);
  transpose_convert_kernel<<<dim3(4, 4), 256, 0, stream>>>(W1, W1t, 128, 128);
  transpose_convert_kernel<<<dim3(2, 4), 256, 0, stream>>>(W2, W2t, 128, 64);

  // 2) gated = sigmoid(x@gate_W+bg) * (x@mark_W+bm)   [fused dual WMMA GEMM]
  gemm_dual_gated_kernel<<<dim3(Cn / 64, (Bn * Tn) / 64), 256, 0, stream>>>(
      xb, markWt, gateWt, mark_b, gate_b, gated, Bn * Tn, Cn, Cn);

  // 3) per-chunk sums, 4) exclusive carry + LN
  chunk_sums_kernel<<<dim3(NCn, Bn), 256, 0, stream>>>(gated, csums);
  carry_ln_kernel<<<dim3(Hn, Bn), 64, 0, stream>>>(csums, carry_g, carry_b, ncarr);

  // 5) cards scan + LN, build combined=[x_heads|cards] bf16
  cards_combined_kernel<<<dim3(NCn, Hn, Bn), 32, 0, stream>>>(
      gated, ncarr, x, card_g, card_b, combined);

  // 6) h = gelu(combined @ W1 + b1)
  gemm_bf16_kernel<EPI_GELU_BF16><<<dim3(2, MH / 64), 256, 0, stream>>>(
      combined, W1t, b1, hbuf, (int)MH, 128, 128);

  // 7) head_out = h @ W2 + b2, fused transpose to (B,T,C)
  gemm_bf16_kernel<EPI_HEADOUT_BF16><<<dim3(1, MH / 64), 256, 0, stream>>>(
      hbuf, W2t, b2, head_out, (int)MH, 64, 128);

  // 8) proj = head_out @ proj_W + proj_b (f32)
  gemm_bf16_kernel<EPI_F32><<<dim3(Cn / 64, (Bn * Tn) / 64), 256, 0, stream>>>(
      head_out, projWt, proj_b, proj_out, Bn * Tn, Cn, Cn);

  // 9) out = x + LN(proj)
  ln_residual_kernel<<<Bn * Tn, 256, 0, stream>>>(x, proj_out, ln_g, ln_b, out);
}
